// FusedLlamaLowBitDecoderlayer_58884001628620
// MI455X (gfx1250) — compile-verified, matrix operand order checked by reading, not run on hardware
//
#include <hip/hip_runtime.h>
#include <hip/hip_bf16.h>

// ---------------------------------------------------------------------------
// Model dims
// ---------------------------------------------------------------------------
constexpr int S     = 1024;
constexpr int H     = 2048;
constexpr int INTER = 5632;
constexpr int NH    = 32;
constexpr int HD    = 64;     // head dim

typedef _Float16 v8h  __attribute__((ext_vector_type(8)));
typedef _Float16 v16h __attribute__((ext_vector_type(16)));
typedef float    v8f  __attribute__((ext_vector_type(8)));
typedef int      v4i  __attribute__((ext_vector_type(4)));

// ---------------------------------------------------------------------------
// Fragment helpers (CDNA5 wave32 WMMA layouts, 16x16x32 f16)
// A (16x32): lane<16 holds row M=lane, K {0..7,16..23}; lane>=16 holds
// K {8..15,24..31}. B (32x16) mirrors with lane = column.  Both reduce to two
// contiguous 8xf16 chunks at +0 and +16 elements.
// ---------------------------------------------------------------------------
__device__ __forceinline__ v16h combine16(v8h lo, v8h hi) {
  v16h r;
#pragma unroll
  for (int i = 0; i < 8; ++i) { r[i] = lo[i]; r[i + 8] = hi[i]; }
  return r;
}

__device__ __forceinline__ v16h load_a16(const _Float16* p) {
  return combine16(*(const v8h*)(p), *(const v8h*)(p + 16));
}

__device__ __forceinline__ v16h load_a16_lds(const _Float16* p) {
  v16h r;
#pragma unroll
  for (int i = 0; i < 8; ++i) { r[i] = p[i]; r[i + 8] = p[16 + i]; }
  return r;
}

__device__ __forceinline__ v8f wmma16(v16h a, v16h b, v8f c) {
  return __builtin_amdgcn_wmma_f32_16x16x32_f16(false, a, false, b, (short)0, c,
                                                false, false);
}

// ---------------------------------------------------------------------------
// One software-pipeline stage of the dequant GEMM: raw int32 weights for two
// B fragments + f16 A chunks for two row tiles.
// ---------------------------------------------------------------------------
struct Stage {
  v4i w0[4];
  v4i w1[4];
  v8h a[4];
};

__device__ __forceinline__ void load_wfrag(v4i w[4], const int* p) {
  w[0] = __builtin_nontemporal_load((const v4i*)(p));
  w[1] = __builtin_nontemporal_load((const v4i*)(p + 4));
  w[2] = __builtin_nontemporal_load((const v4i*)(p + 16));
  w[3] = __builtin_nontemporal_load((const v4i*)(p + 20));
}

__device__ __forceinline__ void stage_load(Stage& s,
                                           const int* wp0, const int* wp1,
                                           const _Float16* ap0,
                                           const _Float16* ap1, int k) {
  load_wfrag(s.w0, wp0 + k);
  load_wfrag(s.w1, wp1 + k);
  s.a[0] = *(const v8h*)(ap0 + k);
  s.a[1] = *(const v8h*)(ap0 + k + 16);
  s.a[2] = *(const v8h*)(ap1 + k);
  s.a[3] = *(const v8h*)(ap1 + k + 16);
}

__device__ __forceinline__ v16h cvt_b(const v4i w[4]) {
  v16h r;
#pragma unroll
  for (int i = 0; i < 4; ++i) {
    r[i]      = (_Float16)w[0][i];
    r[i + 4]  = (_Float16)w[1][i];
    r[i + 8]  = (_Float16)w[2][i];
    r[i + 12] = (_Float16)w[3][i];
  }
  return r;
}

__device__ __forceinline__ void stage_compute(const Stage& s, v8f acc0[2],
                                              v8f acc1[2]) {
  const v16h a0 = combine16(s.a[0], s.a[1]);
  const v16h a1 = combine16(s.a[2], s.a[3]);
  const v16h b0 = cvt_b(s.w0);
  const v16h b1 = cvt_b(s.w1);
  acc0[0] = wmma16(a0, b0, acc0[0]);
  acc1[0] = wmma16(a1, b0, acc1[0]);
  acc0[1] = wmma16(a0, b1, acc0[1]);
  acc1[1] = wmma16(a1, b1, acc1[1]);
}

// ---------------------------------------------------------------------------
// GEMM: out[M,N] (f32) = A[M,K] (f16) x W[N,K]^T (int32-held int8, dequant)
//       * scale[N] (+ optional residual).
// One wave -> 32(M) x 32(N) tile.  Two ping-pong stages, 2x-unrolled K loop
// (K is always a multiple of 64): each stage is reloaded (branchless, offset
// clamped to K-32) immediately after its WMMAs consume it, so one stage of
// non-temporal loads is always in flight with no register copies.
// On top of the 1-deep register pipeline, global_prefetch_b8 pulls weight
// cachelines 4 iterations (1 KB/row) ahead -- latency coverage with zero
// VGPR destinations or counter tracking.
// grid = (M/32, N/128), block = 128 (4 waves).
// ---------------------------------------------------------------------------
__global__ __launch_bounds__(128, 1)
void gemm_f16_i8(const _Float16* __restrict__ A,
                 const int* __restrict__ W,
                 const float* __restrict__ scale,
                 const float* __restrict__ res,
                 float* __restrict__ out,
                 int N, int K) {
  const int lane  = threadIdx.x & 31;
  const int wave  = threadIdx.x >> 5;
  const int hsel  = lane >> 4;
  const int lm    = lane & 15;
  const int mbase = blockIdx.x << 5;                  // 32 rows per wave
  const int nbase = (blockIdx.y * 4 + wave) << 5;     // 32 cols per wave

  const _Float16* aptr0 = A + (size_t)(mbase + lm) * K + (hsel ? 8 : 0);
  const _Float16* aptr1 = aptr0 + (size_t)16 * K;
  const int* wptr0 = W + (size_t)(nbase + lm) * K + (hsel ? 8 : 0);
  const int* wptr1 = wptr0 + (size_t)16 * K;

  v8f acc0[2] = {};   // rows mbase..mbase+15, frag 0/1
  v8f acc1[2] = {};   // rows mbase+16..mbase+31

  Stage s0, s1;
  stage_load(s0, wptr0, wptr1, aptr0, aptr1, 0);
  stage_load(s1, wptr0, wptr1, aptr0, aptr1, 32);

  const int kclamp = K - 32;
  for (int k0 = 0; k0 < K; k0 += 64) {
    // deep prefetch: 4 iterations ahead of the register pipeline
    const int kpf = (k0 + 256 < K) ? (k0 + 256) : kclamp;
    __builtin_prefetch(wptr0 + kpf, 0, 1);
    __builtin_prefetch(wptr1 + kpf, 0, 1);

    stage_compute(s0, acc0, acc1);
    const int kp0 = (k0 + 64 < K) ? (k0 + 64) : kclamp;   // branchless clamp
    stage_load(s0, wptr0, wptr1, aptr0, aptr1, kp0);
    stage_compute(s1, acc0, acc1);
    const int kp1 = (k0 + 96 < K) ? (k0 + 96) : kclamp;
    stage_load(s1, wptr0, wptr1, aptr0, aptr1, kp1);
  }

#pragma unroll
  for (int f = 0; f < 2; ++f) {
    const int n    = nbase + f * 16 + lm;
    const float sc = scale[n];
#pragma unroll
    for (int r = 0; r < 8; ++r) {
      const int row0  = mbase + r + (hsel ? 8 : 0);
      const size_t i0 = (size_t)row0 * N + n;
      const size_t i1 = (size_t)(row0 + 16) * N + n;
      float v0 = acc0[f][r] * sc;
      float v1 = acc1[f][r] * sc;
      if (res) { v0 += res[i0]; v1 += res[i1]; }
      out[i0] = v0;
      out[i1] = v1;
    }
  }
}

// ---------------------------------------------------------------------------
// RMSNorm: one block per row; f32 in, f16 out.
// ---------------------------------------------------------------------------
__global__ void rmsnorm_kernel(const float* __restrict__ x,
                               const float* __restrict__ w,
                               _Float16* __restrict__ out) {
  __shared__ float red[256];
  const int row = blockIdx.x;
  const float* xr = x + (size_t)row * H;
  float ss = 0.f;
  for (int i = threadIdx.x; i < H; i += 256) { float v = xr[i]; ss += v * v; }
  red[threadIdx.x] = ss;
  __syncthreads();
  for (int s = 128; s > 0; s >>= 1) {
    if (threadIdx.x < s) red[threadIdx.x] += red[threadIdx.x + s];
    __syncthreads();
  }
  const float inv = rsqrtf(red[0] / (float)H + 1e-5f);
  for (int i = threadIdx.x; i < H; i += 256)
    out[(size_t)row * H + i] = (_Float16)(w[i] * xr[i] * inv);
}

// ---------------------------------------------------------------------------
// RoPE on Q,K (f32 -> f16) + V transpose to [col][S] f16.
// ---------------------------------------------------------------------------
__global__ void rope_vt_kernel(const float* __restrict__ q32,
                               const float* __restrict__ k32,
                               const float* __restrict__ v32,
                               const float* __restrict__ cosb,
                               const float* __restrict__ sinb,
                               const long long* __restrict__ pos,
                               _Float16* __restrict__ q16,
                               _Float16* __restrict__ k16,
                               _Float16* __restrict__ vt) {
  const int idx  = blockIdx.x * blockDim.x + threadIdx.x;  // < S*H
  const int srow = idx >> 11;            // / H
  const int col  = idx & (H - 1);
  const int d    = col & 63;
  const int p    = (int)pos[srow];
  const float c  = cosb[p * HD + d];
  const float sn = sinb[p * HD + d];
  const float qv = q32[idx], kv = k32[idx];
  float qr, kr;
  if (d < 32) { qr = qv * c - q32[idx + 32] * sn; kr = kv * c - k32[idx + 32] * sn; }
  else        { qr = qv * c + q32[idx - 32] * sn; kr = kv * c + k32[idx - 32] * sn; }
  q16[idx] = (_Float16)qr;
  k16[idx] = (_Float16)kr;
  vt[(size_t)col * S + srow] = (_Float16)v32[idx];
}

// ---------------------------------------------------------------------------
// Flash attention: one wave per (head, 16-row Q tile).
// grid = (S/16/4, NH), block = 128 (4 waves).
// P tile round-trips through LDS for C-layout -> A-layout conversion.
// ---------------------------------------------------------------------------
__global__ void attn_kernel(const _Float16* __restrict__ Q,
                            const _Float16* __restrict__ Kf,
                            const _Float16* __restrict__ Vt,
                            const float* __restrict__ mask,
                            _Float16* __restrict__ O) {
  __shared__ __align__(16) _Float16 lds_p[4][16 * 32];
  const int lane  = threadIdx.x & 31;
  const int wave  = threadIdx.x >> 5;
  const int hsel  = lane >> 4;
  const int lm    = lane & 15;
  const int mbase = (blockIdx.x * 4 + wave) * 16;
  const int hoff  = blockIdx.y * HD;

  const v16h aq0 = load_a16(Q + (size_t)(mbase + lm) * H + hoff + (hsel ? 8 : 0));
  const v16h aq1 = load_a16(Q + (size_t)(mbase + lm) * H + hoff + 32 + (hsel ? 8 : 0));

  float mrow[8], lrow[8];
#pragma unroll
  for (int r = 0; r < 8; ++r) { mrow[r] = -3.0e38f; lrow[r] = 0.f; }
  v8f o0 = {}, o1 = {}, o2 = {}, o3 = {};

  for (int j0 = 0; j0 < S; j0 += 32) {
    v8f sc[2];
#pragma unroll
    for (int jt = 0; jt < 2; ++jt) {
      const int kb = j0 + jt * 16;
      const v16h bk0 = load_a16(Kf + (size_t)(kb + lm) * H + hoff + (hsel ? 8 : 0));
      const v16h bk1 = load_a16(Kf + (size_t)(kb + lm) * H + hoff + 32 + (hsel ? 8 : 0));
      v8f c = {};
      c = wmma16(aq0, bk0, c);
      c = wmma16(aq1, bk1, c);
#pragma unroll
      for (int r = 0; r < 8; ++r) {
        const int row = mbase + r + (hsel ? 8 : 0);
        sc[jt][r] = c[r] * 0.125f + mask[(size_t)row * S + kb + lm];
      }
    }
    // online softmax per row (rows live across 16-lane halves)
#pragma unroll
    for (int r = 0; r < 8; ++r) {
      float mx = fmaxf(sc[0][r], sc[1][r]);
      mx = fmaxf(mx, __shfl_xor(mx, 1));
      mx = fmaxf(mx, __shfl_xor(mx, 2));
      mx = fmaxf(mx, __shfl_xor(mx, 4));
      mx = fmaxf(mx, __shfl_xor(mx, 8));
      const float nm = fmaxf(mrow[r], mx);
      const float ef = __expf(mrow[r] - nm);
      mrow[r] = nm;
      lrow[r] *= ef;
      o0[r] *= ef; o1[r] *= ef; o2[r] *= ef; o3[r] *= ef;
      const float p0 = __expf(sc[0][r] - nm);
      const float p1 = __expf(sc[1][r] - nm);
      float rs = p0 + p1;
      rs += __shfl_xor(rs, 1);
      rs += __shfl_xor(rs, 2);
      rs += __shfl_xor(rs, 4);
      rs += __shfl_xor(rs, 8);
      lrow[r] += rs;
      _Float16* lp = &lds_p[wave][(r + (hsel ? 8 : 0)) * 32];
      lp[lm]      = (_Float16)p0;
      lp[16 + lm] = (_Float16)p1;
    }
    // P (16x32) as A-fragment from LDS, V rows as B-fragments
    const v16h ap = load_a16_lds(&lds_p[wave][lm * 32 + (hsel ? 8 : 0)]);
    v16h bv;
    bv = load_a16(Vt + (size_t)(hoff + 0  + lm) * S + j0 + (hsel ? 8 : 0)); o0 = wmma16(ap, bv, o0);
    bv = load_a16(Vt + (size_t)(hoff + 16 + lm) * S + j0 + (hsel ? 8 : 0)); o1 = wmma16(ap, bv, o1);
    bv = load_a16(Vt + (size_t)(hoff + 32 + lm) * S + j0 + (hsel ? 8 : 0)); o2 = wmma16(ap, bv, o2);
    bv = load_a16(Vt + (size_t)(hoff + 48 + lm) * S + j0 + (hsel ? 8 : 0)); o3 = wmma16(ap, bv, o3);
  }

#pragma unroll
  for (int r = 0; r < 8; ++r) {
    const float inv = 1.0f / lrow[r];
    const int row = mbase + r + (hsel ? 8 : 0);
    _Float16* op = O + (size_t)row * H + hoff;
    op[lm]      = (_Float16)(o0[r] * inv);
    op[16 + lm] = (_Float16)(o1[r] * inv);
    op[32 + lm] = (_Float16)(o2[r] * inv);
    op[48 + lm] = (_Float16)(o3[r] * inv);
  }
}

// ---------------------------------------------------------------------------
// silu(gate) * up -> f16
// ---------------------------------------------------------------------------
__global__ void silu_mul_kernel(const float* __restrict__ g,
                                const float* __restrict__ u,
                                _Float16* __restrict__ act) {
  const size_t idx = (size_t)blockIdx.x * blockDim.x + threadIdx.x;
  const float gv = g[idx];
  const float sv = gv / (1.f + __expf(-gv));
  act[idx] = (_Float16)(sv * u[idx]);
}

// ---------------------------------------------------------------------------
// Workspace layout (bytes).  Peak usage 78 MiB with aliasing:
// q/k/v f32 die before gate; q/k/v f16 + attn die before act.
// ---------------------------------------------------------------------------
constexpr size_t MB = 1024ull * 1024ull;
constexpr size_t OFF_XN   = 0;          // f16 [S,H]     4 MiB
constexpr size_t OFF_QF32 = 4 * MB;     // f32 [S,H]     8 MiB
constexpr size_t OFF_KF32 = 12 * MB;
constexpr size_t OFF_VF32 = 20 * MB;
constexpr size_t OFF_QF16 = 28 * MB;    // f16 [S,H]
constexpr size_t OFF_KF16 = 32 * MB;
constexpr size_t OFF_VT   = 36 * MB;    // f16 [H,S] (per-head d-major)
constexpr size_t OFF_ATTN = 40 * MB;    // f16 [S,H]
constexpr size_t OFF_HID  = 44 * MB;    // f32 [S,H]     8 MiB
constexpr size_t OFF_H2   = 52 * MB;    // f16 [S,H]
constexpr size_t OFF_GATE = 4 * MB;     // f32 [S,INTER] 22 MiB  (aliases q/k/v f32)
constexpr size_t OFF_UP   = 56 * MB;    // f32 [S,INTER] 22 MiB  -> end 78 MiB
constexpr size_t OFF_ACT  = 28 * MB;    // f16 [S,INTER] 11 MiB  (aliases q/k f16)

extern "C" void kernel_launch(void* const* d_in, const int* in_sizes, int n_in,
                              void* d_out, int out_size, void* d_ws, size_t ws_size,
                              hipStream_t stream) {
  (void)in_sizes; (void)n_in; (void)out_size; (void)ws_size;
  const float*     hidden = (const float*)d_in[0];
  const float*     mask   = (const float*)d_in[1];
  const long long* pos    = (const long long*)d_in[2];
  const float*     ln0    = (const float*)d_in[3];
  const float*     ln1    = (const float*)d_in[4];
  const float*     cosb   = (const float*)d_in[5];
  const float*     sinb   = (const float*)d_in[6];
  const int*   q_w = (const int*)d_in[7];   const float* q_s = (const float*)d_in[8];
  const int*   k_w = (const int*)d_in[9];   const float* k_s = (const float*)d_in[10];
  const int*   v_w = (const int*)d_in[11];  const float* v_s = (const float*)d_in[12];
  const int*   o_w = (const int*)d_in[13];  const float* o_s = (const float*)d_in[14];
  const int*   g_w = (const int*)d_in[15];  const float* g_s = (const float*)d_in[16];
  const int*   u_w = (const int*)d_in[17];  const float* u_s = (const float*)d_in[18];
  const int*   dnw = (const int*)d_in[19];  const float* dns = (const float*)d_in[20];

  char* ws = (char*)d_ws;
  _Float16* xn   = (_Float16*)(ws + OFF_XN);
  float*    qf32 = (float*)(ws + OFF_QF32);
  float*    kf32 = (float*)(ws + OFF_KF32);
  float*    vf32 = (float*)(ws + OFF_VF32);
  _Float16* qf16 = (_Float16*)(ws + OFF_QF16);
  _Float16* kf16 = (_Float16*)(ws + OFF_KF16);
  _Float16* vt   = (_Float16*)(ws + OFF_VT);
  _Float16* attn = (_Float16*)(ws + OFF_ATTN);
  float*    hid  = (float*)(ws + OFF_HID);
  _Float16* h2   = (_Float16*)(ws + OFF_H2);
  float*    gate = (float*)(ws + OFF_GATE);
  float*    up   = (float*)(ws + OFF_UP);
  _Float16* act  = (_Float16*)(ws + OFF_ACT);
  float*    out  = (float*)d_out;

  const dim3 blk(128);
  const dim3 gH(S / 32, H / 128);       // 32 x 16
  const dim3 gI(S / 32, INTER / 128);   // 32 x 44

  // 1. RMSNorm -> f16 activations
  rmsnorm_kernel<<<S, 256, 0, stream>>>(hidden, ln0, xn);
  // 2. Q,K,V projections (WMMA)
  gemm_f16_i8<<<gH, blk, 0, stream>>>(xn, q_w, q_s, nullptr, qf32, H, H);
  gemm_f16_i8<<<gH, blk, 0, stream>>>(xn, k_w, k_s, nullptr, kf32, H, H);
  gemm_f16_i8<<<gH, blk, 0, stream>>>(xn, v_w, v_s, nullptr, vf32, H, H);
  // 3. RoPE + V transpose
  rope_vt_kernel<<<(S * H) / 256, 256, 0, stream>>>(qf32, kf32, vf32, cosb, sinb,
                                                    pos, qf16, kf16, vt);
  // 4. Flash attention (WMMA)
  attn_kernel<<<dim3(S / 64, NH), blk, 0, stream>>>(qf16, kf16, vt, mask, attn);
  // 5. O projection + residual
  gemm_f16_i8<<<gH, blk, 0, stream>>>(attn, o_w, o_s, hidden, hid, H, H);
  // 6. RMSNorm
  rmsnorm_kernel<<<S, 256, 0, stream>>>(hid, ln1, h2);
  // 7. gate / up projections
  gemm_f16_i8<<<gI, blk, 0, stream>>>(h2, g_w, g_s, nullptr, gate, INTER, H);
  gemm_f16_i8<<<gI, blk, 0, stream>>>(h2, u_w, u_s, nullptr, up, INTER, H);
  // 8. silu(gate)*up
  silu_mul_kernel<<<(S * INTER) / 256, 256, 0, stream>>>(gate, up, act);
  // 9. down projection + residual -> output
  gemm_f16_i8<<<gH, blk, 0, stream>>>(act, dnw, dns, hid, out, H, INTER);
}
